// HConstructor_for_visual_7121055777191
// MI455X (gfx1250) — matrix operand order repaired and checked by reading, not compiled, not run on hardware
//
#include <hip/hip_runtime.h>
#include <hip/hip_bf16.h>

typedef unsigned short u16;
typedef __attribute__((ext_vector_type(16))) __bf16 v16bf;
typedef __attribute__((ext_vector_type(8)))  __bf16 v8bf;
typedef __attribute__((ext_vector_type(8)))  float  v8f;
typedef __attribute__((ext_vector_type(16))) short  v16s;

#define N_NODES 50000
#define DIM     256
#define NEDGE   512
#define MROWS   (4 * N_NODES)   // 200000
#define KPAD    50016           // N_NODES padded to multiple of 32
#define SPLITK  64

__device__ __forceinline__ u16 f2bf(float f) {
  unsigned u = __float_as_uint(f);
  u += 0x7FFFu + ((u >> 16) & 1u);   // round-to-nearest-even
  return (u16)(u >> 16);
}

union Frag { v16bf v; v8bf h[2]; v16s s; };

// epilogue modes (compile-time)
#define EPI_BF16       0
#define EPI_BF16_RELU  1
#define EPI_F32_SCALE  3
#define EPI_F32_PART   4
#define EPI_ARGMAX     5

// C = relu?(A[M,K]) * B[K,Nn] + bias, bf16 inputs, f32 accum via v_wmma_f32_16x16x32_bf16.
// Block tile 128x128x32, 256 threads = 8 waves (4 along M x 2 along N), 32x64 per wave.
// A tile staged via GLOBAL_LOAD_ASYNC_TO_LDS_B128 (ASYNCcnt); relu applied in-register
// on the bf16 fragments (packed i16 max, valid on bf16 bit patterns).
template <int MODE, bool RELUA>
__global__ __launch_bounds__(256)
void gemm_bf16_wmma(const u16* __restrict__ A, long long lda,
                    const u16* __restrict__ B, long long ldb,
                    const float* __restrict__ bias,
                    int M, int K, int kChunkSteps, float alpha,
                    u16* __restrict__ outBf, float* __restrict__ outF,
                    long long ldc, long long partStride,
                    float* __restrict__ argval, int* __restrict__ argidx)
{
  __shared__ union {
    struct { u16 a[128 * 40]; u16 b[128 * 40]; } t;   // LDS tiles, stride 40 (bank-conflict pad)
    struct { float v[128 * 32]; int i[128 * 32]; } am; // argmax scratch
  } sm;

  const int tid  = threadIdx.x;
  const int lane = tid & 31;
  const int w    = tid >> 5;
  const int wm   = w & 3;        // wave row 0..3
  const int wn   = w >> 2;       // wave col 0..1
  const int l15  = lane & 15;
  const int lh   = lane >> 4;

  const int blockN = blockIdx.x * 128;
  const int blockM = blockIdx.y * 128;

  int kStart = blockIdx.z * kChunkSteps * 32;
  int kEnd   = kStart + kChunkSteps * 32;
  if (kEnd > K) kEnd = K;

  v8f acc[2][4];
#pragma unroll
  for (int i = 0; i < 2; ++i)
#pragma unroll
    for (int j = 0; j < 4; ++j)
#pragma unroll
      for (int r = 0; r < 8; ++r) acc[i][j][r] = 0.f;

  // A staging: 2 threads per row, 16 bf16 (32B) each -> async copy direct to LDS
  const int aRow = tid >> 1;
  const int aCol = (tid & 1) * 16;
  const long long aGRow = (long long)blockM + aRow;
  const bool aValid = (aGRow < (long long)M);
  const u16* aPtr = A + aGRow * lda + aCol;
  const unsigned aLds = (unsigned)(uintptr_t)&sm.t.a[aRow * 40 + aCol];

  // B staging: thread covers 2 consecutive k rows x 8 columns, stored transposed [n][k]
  const int bCol = (tid & 15) * 8;
  const int bK   = (tid >> 4) * 2;
  const u16* bPtr = B + (long long)bK * ldb + blockN + bCol;

  for (int k0 = kStart; k0 < kEnd; k0 += 32) {
    __syncthreads();
    if (aValid) {
      unsigned long long ga = (unsigned long long)(uintptr_t)(aPtr + k0);
      // same inst_offset applies to both global source and LDS dest (ISA 15.18.3)
      asm volatile("global_load_async_to_lds_b128 %0, %1, off\n\t"
                   "global_load_async_to_lds_b128 %0, %1, off offset:16"
                   :: "v"(aLds), "v"(ga) : "memory");
    }
    {
      uint4 r0 = *(const uint4*)(bPtr + (long long)k0 * ldb);
      uint4 r1 = *(const uint4*)(bPtr + (long long)(k0 + 1) * ldb);
      const u16* s0 = (const u16*)&r0;
      const u16* s1 = (const u16*)&r1;
#pragma unroll
      for (int i = 0; i < 8; ++i) {
        unsigned pack = (unsigned)s0[i] | ((unsigned)s1[i] << 16);
        *(unsigned*)&sm.t.b[(bCol + i) * 40 + bK] = pack;
      }
    }
    asm volatile("s_wait_asynccnt 0" ::: "memory");
    __syncthreads();

    Frag fa[2], fb[4];
#pragma unroll
    for (int tm = 0; tm < 2; ++tm) {
      int base = (wm * 32 + tm * 16 + l15) * 40 + lh * 8;
      fa[tm].h[0] = *(const v8bf*)&sm.t.a[base];       // K: lh*8 .. +7
      fa[tm].h[1] = *(const v8bf*)&sm.t.a[base + 16];  // K: lh*8+16 .. +23
      if (RELUA) {
        v16s z = {};
        fa[tm].s = __builtin_elementwise_max(fa[tm].s, z);  // bf16 relu on bit patterns
      }
    }
#pragma unroll
    for (int tn = 0; tn < 4; ++tn) {
      int base = (wn * 64 + tn * 16 + l15) * 40 + lh * 8;
      fb[tn].h[0] = *(const v8bf*)&sm.t.b[base];
      fb[tn].h[1] = *(const v8bf*)&sm.t.b[base + 16];
    }
#pragma unroll
    for (int tm = 0; tm < 2; ++tm)
#pragma unroll
      for (int tn = 0; tn < 4; ++tn)
        acc[tm][tn] = __builtin_amdgcn_wmma_f32_16x16x32_bf16(
            false, fa[tm].v, false, fb[tn].v, (short)0, acc[tm][tn], false, false);
  }

  if (MODE == EPI_ARGMAX) {
    __syncthreads();   // LDS union reuse
#pragma unroll
    for (int tm = 0; tm < 2; ++tm) {
#pragma unroll
      for (int r = 0; r < 8; ++r) {
        float bv = -3.4e38f; int bn = 0x7FFFFFFF;
#pragma unroll
        for (int tn = 0; tn < 4; ++tn) {
          int nG = blockN + wn * 64 + tn * 16 + l15;
          float v = acc[tm][tn][r] + (bias ? bias[nG] : 0.f);
          if (v > bv || (v == bv && nG < bn)) { bv = v; bn = nG; }
        }
        int rowL = wm * 32 + tm * 16 + lh * 8 + r;
        int slot = wn * 16 + l15;
        sm.am.v[rowL * 32 + slot] = bv;
        sm.am.i[rowL * 32 + slot] = bn;
      }
    }
    __syncthreads();
    if (tid < 128) {
      long long mG = (long long)blockM + tid;
      if (mG < (long long)M) {
        float bv = sm.am.v[tid * 32]; int bn = sm.am.i[tid * 32];
        for (int s = 1; s < 32; ++s) {
          float v = sm.am.v[tid * 32 + s]; int n2 = sm.am.i[tid * 32 + s];
          if (v > bv || (v == bv && n2 < bn)) { bv = v; bn = n2; }
        }
        argval[mG * gridDim.x + blockIdx.x] = bv;
        argidx[mG * gridDim.x + blockIdx.x] = bn;
      }
    }
    return;
  }

#pragma unroll
  for (int tm = 0; tm < 2; ++tm)
#pragma unroll
    for (int tn = 0; tn < 4; ++tn) {
      int nG = blockN + wn * 64 + tn * 16 + l15;
      float bv = bias ? bias[nG] : 0.f;
#pragma unroll
      for (int r = 0; r < 8; ++r) {
        long long mG = (long long)blockM + wm * 32 + tm * 16 + lh * 8 + r;
        if (mG >= (long long)M) continue;
        float v = acc[tm][tn][r] + bv;
        long long off = mG * ldc + nG;
        if      (MODE == EPI_BF16)      outBf[off] = f2bf(v);
        else if (MODE == EPI_BF16_RELU) outBf[off] = f2bf(fmaxf(v, 0.f));
        else if (MODE == EPI_F32_SCALE) outF[off]  = v * alpha;
        else                            outF[(long long)blockIdx.z * partStride + off] = v;
      }
    }
}

// ---------------- small helper kernels ----------------

__global__ void f32_to_bf16_kernel(const float* __restrict__ in, u16* __restrict__ out, long long n) {
  long long i = (long long)blockIdx.x * blockDim.x + threadIdx.x;
  if (i < n) out[i] = f2bf(in[i]);
}

__global__ void combine_argmax_kernel(const float* __restrict__ argval,
                                      const int* __restrict__ argidx,
                                      int* __restrict__ classes, int nTiles, int M) {
  int r = blockIdx.x * blockDim.x + threadIdx.x;
  if (r >= M) return;
  long long base = (long long)r * nTiles;
  float bv = argval[base]; int bn = argidx[base];
  for (int t = 1; t < nTiles; ++t) {
    float v = argval[base + t]; int n2 = argidx[base + t];
    if (v > bv || (v == bv && n2 < bn)) { bv = v; bn = n2; }
  }
  classes[r] = bn;
}

__global__ void node_stats_kernel(const int* __restrict__ classes,
                                  int* __restrict__ hist, u16* __restrict__ maskT) {
  int n = blockIdx.x * blockDim.x + threadIdx.x;
  if (n >= N_NODES) return;
  int es[4];
  es[0] = classes[n];               es[1] = classes[N_NODES + n];
  es[2] = classes[2 * N_NODES + n]; es[3] = classes[3 * N_NODES + n];
#pragma unroll
  for (int i = 0; i < 4; ++i) {
    bool dup = false;
    for (int j = 0; j < i; ++j) if (es[j] == es[i]) dup = true;
    if (dup) continue;
    int c = 1;
    for (int j = i + 1; j < 4; ++j) if (es[j] == es[i]) c++;
    atomicAdd(&hist[es[i] * 5 + c], 1);
    maskT[(long long)es[i] * KPAD + n] = 0x3F80u;  // bf16 1.0
  }
}

__global__ void col_stats_kernel(const int* __restrict__ hist,
                                 float* __restrict__ colmax, float* __restrict__ colinv,
                                 float* __restrict__ colbase) {
  int e = blockIdx.x * blockDim.x + threadIdx.x;
  if (e >= NEDGE) return;
  int h1 = hist[e * 5 + 1], h2 = hist[e * 5 + 2], h3 = hist[e * 5 + 3], h4 = hist[e * 5 + 4];
  int mx = h4 ? 4 : (h3 ? 3 : (h2 ? 2 : (h1 ? 1 : 0)));
  int npos = h1 + h2 + h3 + h4;
  float fm = (float)mx;
  float sum = (float)(N_NODES - npos) * expf(-fm)
            + h1 * expf(1.f - fm) + h2 * expf(2.f - fm)
            + h3 * expf(3.f - fm) + h4 * expf(4.f - fm);
  colmax[e]  = fm;
  colinv[e]  = 1.f / sum;
  colbase[e] = expf(-fm) / sum;
}

__global__ void h_fill_kernel(float* __restrict__ H, const float* __restrict__ colbase) {
  long long i = (long long)blockIdx.x * blockDim.x + threadIdx.x;
  if (i < (long long)N_NODES * NEDGE) H[i] = colbase[(int)(i % NEDGE)];
}

__global__ void h_set_kernel(const int* __restrict__ classes, const float* __restrict__ colmax,
                             const float* __restrict__ colinv, float* __restrict__ H) {
  int n = blockIdx.x * blockDim.x + threadIdx.x;
  if (n >= N_NODES) return;
  int es[4];
  es[0] = classes[n];               es[1] = classes[N_NODES + n];
  es[2] = classes[2 * N_NODES + n]; es[3] = classes[3 * N_NODES + n];
#pragma unroll
  for (int i = 0; i < 4; ++i) {
    bool dup = false;
    for (int j = 0; j < i; ++j) if (es[j] == es[i]) dup = true;
    if (dup) continue;
    int c = 1;
    for (int j = i + 1; j < 4; ++j) if (es[j] == es[i]) c++;
    H[(long long)n * NEDGE + es[i]] = expf((float)c - colmax[es[i]]) * colinv[es[i]];
  }
}

__global__ void pool_reduce_kernel(const float* __restrict__ part, float* __restrict__ hf) {
  int i = blockIdx.x * blockDim.x + threadIdx.x;
  if (i >= NEDGE * DIM) return;
  float s = 0.f;
  for (int z = 0; z < SPLITK; ++z) s += part[(long long)z * (NEDGE * DIM) + i];
  hf[i] = s;
}

__global__ void hf_transpose_bf16_kernel(const float* __restrict__ hf, u16* __restrict__ hfT) {
  int i = blockIdx.x * blockDim.x + threadIdx.x;
  if (i >= NEDGE * DIM) return;
  int e = i / DIM, d = i % DIM;
  hfT[(long long)d * NEDGE + e] = f2bf(hf[i]);
}

// ---------------- host launch ----------------

template <int MODE, bool RELUA>
static void launch_gemm(hipStream_t s, const u16* A, long long lda, const u16* B, long long ldb,
                        const float* bias, int M, int Nn, int K, int zChunks, int kChunkSteps,
                        float alpha, u16* outBf, float* outF,
                        long long ldc, long long partStride, float* argval, int* argidx) {
  dim3 grid(Nn / 128, (M + 127) / 128, zChunks);
  gemm_bf16_wmma<MODE, RELUA><<<grid, 256, 0, s>>>(A, lda, B, ldb, bias, M, K, kChunkSteps,
                                                   alpha, outBf, outF, ldc, partStride,
                                                   argval, argidx);
}

extern "C" void kernel_launch(void* const* d_in, const int* in_sizes, int n_in,
                              void* d_out, int out_size, void* d_ws, size_t ws_size,
                              hipStream_t stream) {
  (void)in_sizes; (void)n_in; (void)out_size; (void)ws_size;
  const float* features = (const float*)d_in[0];
  const float* W_lin    = (const float*)d_in[1];
  const float* b_lin    = (const float*)d_in[2];
  const float* W0       = (const float*)d_in[3];
  const float* b0       = (const float*)d_in[4];
  const float* W1       = (const float*)d_in[5];
  const float* b1       = (const float*)d_in[6];
  const float* We       = (const float*)d_in[7];
  const float* be       = (const float*)d_in[8];

  float* H    = (float*)d_out;                       // [N, E]
  float* hf   = H + (long long)N_NODES * NEDGE;      // [E, D]
  float* dots = hf + (long long)NEDGE * DIM;         // [4N, E]

  char* p = (char*)d_ws;
  auto carve = [&](size_t bytes) -> void* {
    void* r = (void*)p;
    p += (bytes + 255) & ~(size_t)255;
    return r;
  };
  u16*   AFbf   = (u16*)carve((size_t)MROWS * DIM * 2);       // all_features bf16
  u16*   Xbf    = (u16*)carve((size_t)MROWS * DIM * 2);       // relu(x) bf16
  u16*   AF2bf  = (u16*)carve((size_t)MROWS * DIM * 2);       // all_features2 bf16
  u16*   maskT  = (u16*)carve((size_t)NEDGE * KPAD * 2);      // mask^T bf16, zero-padded
  u16*   Wlinbf = (u16*)carve((size_t)3 * DIM * DIM * 2);
  u16*   W0bf   = (u16*)carve((size_t)DIM * DIM * 2);
  u16*   W1bf   = (u16*)carve((size_t)DIM * DIM * 2);
  u16*   Webf   = (u16*)carve((size_t)DIM * NEDGE * 2);
  u16*   hfT    = (u16*)carve((size_t)DIM * NEDGE * 2);
  float* argval = (float*)carve((size_t)MROWS * 4 * 4);
  int*   argidx = (int*)carve((size_t)MROWS * 4 * 4);
  int*   classes= (int*)carve((size_t)MROWS * 4);
  int*   hist   = (int*)carve((size_t)NEDGE * 5 * 4);
  float* colmax = (float*)carve((size_t)NEDGE * 4);
  float* colinv = (float*)carve((size_t)NEDGE * 4);
  float* colbase= (float*)carve((size_t)NEDGE * 4);
  float* poolpart=(float*)carve((size_t)SPLITK * NEDGE * DIM * 4);

  hipMemsetAsync(maskT, 0, (size_t)NEDGE * KPAD * 2, stream);
  hipMemsetAsync(hist, 0, (size_t)NEDGE * 5 * 4, stream);

  // bf16 conversions: weights + features (features land in rows 0..N-1 of AFbf)
  {
    long long n;
    n = 3LL * DIM * DIM;  f32_to_bf16_kernel<<<(unsigned)((n + 255) / 256), 256, 0, stream>>>(W_lin, Wlinbf, n);
    n = (long long)DIM * DIM; f32_to_bf16_kernel<<<(unsigned)((n + 255) / 256), 256, 0, stream>>>(W0, W0bf, n);
    f32_to_bf16_kernel<<<(unsigned)((n + 255) / 256), 256, 0, stream>>>(W1, W1bf, n);
    n = (long long)DIM * NEDGE; f32_to_bf16_kernel<<<(unsigned)((n + 255) / 256), 256, 0, stream>>>(We, Webf, n);
    n = (long long)N_NODES * DIM; f32_to_bf16_kernel<<<(unsigned)((n + 255) / 256), 256, 0, stream>>>(features, AFbf, n);
  }

  // 1) view projections: rows (t+1)*N of all_features
  for (int t = 0; t < 3; ++t)
    launch_gemm<EPI_BF16, false>(stream, AFbf, DIM, Wlinbf + (size_t)t * DIM * DIM, DIM,
                                 b_lin + t * DIM, N_NODES, DIM, DIM, 1, DIM / 32, 1.f,
                                 AFbf + (size_t)(t + 1) * N_NODES * DIM, nullptr, DIM, 0,
                                 nullptr, nullptr);

  // 2) x = relu(all_features) @ W0 + b0, store relu(x)
  launch_gemm<EPI_BF16_RELU, true>(stream, AFbf, DIM, W0bf, DIM, b0, MROWS, DIM, DIM, 1,
                                   DIM / 32, 1.f, Xbf, nullptr, DIM, 0, nullptr, nullptr);

  // 3) all_features2 = relu(x) @ W1 + b1
  launch_gemm<EPI_BF16, false>(stream, Xbf, DIM, W1bf, DIM, b1, MROWS, DIM, DIM, 1, DIM / 32,
                               1.f, AF2bf, nullptr, DIM, 0, nullptr, nullptr);

  // 4) logits = relu(all_features2) @ We + be, fused per-tile row argmax
  launch_gemm<EPI_ARGMAX, true>(stream, AF2bf, DIM, Webf, NEDGE, be, MROWS, NEDGE, DIM, 1,
                                DIM / 32, 1.f, nullptr, nullptr, NEDGE, 0, argval, argidx);

  // 5) combine partial argmaxes -> classes
  combine_argmax_kernel<<<(MROWS + 255) / 256, 256, 0, stream>>>(argval, argidx, classes,
                                                                 NEDGE / 128, MROWS);

  // 6) per-node dedupe -> count histogram (int atomics) + mask^T
  node_stats_kernel<<<(N_NODES + 255) / 256, 256, 0, stream>>>(classes, hist, maskT);

  // 7) per-column softmax stats, then H
  col_stats_kernel<<<(NEDGE + 255) / 256, 256, 0, stream>>>(hist, colmax, colinv, colbase);
  {
    long long n = (long long)N_NODES * NEDGE;
    h_fill_kernel<<<(unsigned)((n + 255) / 256), 256, 0, stream>>>(H, colbase);
  }
  h_set_kernel<<<(N_NODES + 255) / 256, 256, 0, stream>>>(classes, colmax, colinv, H);

  // 8) hyperedge pooling: hf[E,D] = mask^T @ all_features2[:N]  (split-K WMMA)
  launch_gemm<EPI_F32_PART, false>(stream, maskT, KPAD, AF2bf, DIM, nullptr, NEDGE, DIM, KPAD,
                                   SPLITK, (1563 + SPLITK - 1) / SPLITK, 1.f, nullptr, poolpart,
                                   DIM, (long long)NEDGE * DIM, nullptr, nullptr);
  pool_reduce_kernel<<<(NEDGE * DIM + 255) / 256, 256, 0, stream>>>(poolpart, hf);

  // 9) dots = all_features @ hf^T * D^-0.5
  hf_transpose_bf16_kernel<<<(NEDGE * DIM + 255) / 256, 256, 0, stream>>>(hf, hfT);
  launch_gemm<EPI_F32_SCALE, false>(stream, AFbf, DIM, hfT, NEDGE, nullptr, MROWS, NEDGE, DIM,
                                    1, DIM / 32, 0.0625f, nullptr, dots, NEDGE, 0,
                                    nullptr, nullptr);
}
